// OLMoE1B7B0924Synapse_8220567404859
// MI455X (gfx1250) — compile-verified, matrix operand order checked by reading, not run on hardware
//
#include <hip/hip_runtime.h>
#include <hip/hip_bf16.h>

#define D_MODEL 2048
#define NHEADS 16
#define HDIM 128
#define NEXPERTS 64
#define NTOPK 8
#define FFDIM 1024
#define SEQ 1024
#define EPSF 1e-5f

typedef __bf16 v16bf __attribute__((ext_vector_type(16)));
typedef __bf16 v2bf  __attribute__((ext_vector_type(2)));
typedef float  v8f   __attribute__((ext_vector_type(8)));

struct Frag { union { v16bf v; uint4 q[2]; }; };

// Packed fp32->bf16 conversion: force one v_cvt_pk_bf16_f32 per float pair.
// (clang's scalar (__bf16) casts lower to cvt_pk with an undef second operand
// plus v_mov_b16 repacking; direct asm emits the true packed form.)
__device__ __forceinline__ unsigned pk2(float a, float b) {
  unsigned r;
  asm("v_cvt_pk_bf16_f32 %0, %1, %2" : "=v"(r) : "v"(a), "v"(b));
  return r;
}
__device__ __forceinline__ unsigned short bf16bits(float a) {
  return __builtin_bit_cast(unsigned short, (__bf16)a);
}

// A/B fragment loader from LDS tile. Per CDNA5 WMMA 16-bit A layout:
// lane holds row (lane&15); K elements e: K = 16*(e/8) + g*8 + (e%8), g = lane>>4.
// => two contiguous 16B chunks at (kbase + g*8) and (kbase + 16 + g*8).
__device__ __forceinline__ void ld_frag(const unsigned short* base, int row,
                                        int stride, int kbase, Frag& f) {
  int g = (threadIdx.x >> 4) & 1;
  const unsigned short* p = base + row * stride + kbase + g * 8;
  f.q[0] = *(const uint4*)p;
  f.q[1] = *(const uint4*)(p + 16);
}

__device__ __forceinline__ v8f wmma_bf16(const Frag& a, const Frag& b, v8f c) {
  return __builtin_amdgcn_wmma_f32_16x16x32_bf16(false, a.v, false, b.v,
                                                 (short)0, c, false, false);
}

// ---------------------------------------------------------------- RMSNorm
__global__ __launch_bounds__(256) void rmsnorm_kernel(
    const float* __restrict__ in, const float* __restrict__ w,
    float* __restrict__ out) {
  int row = blockIdx.x;
  const float* r = in + (size_t)row * D_MODEL;
  int base = threadIdx.x * 8;
  const float4* p = (const float4*)(r + base);
  float4 a = p[0], b = p[1];
  float ss = a.x*a.x + a.y*a.y + a.z*a.z + a.w*a.w
           + b.x*b.x + b.y*b.y + b.z*b.z + b.w*b.w;
  #pragma unroll
  for (int m = 16; m >= 1; m >>= 1) ss += __shfl_xor(ss, m, 32);
  __shared__ float sred[8];
  if ((threadIdx.x & 31) == 0) sred[threadIdx.x >> 5] = ss;
  __syncthreads();
  float tot = sred[0]+sred[1]+sred[2]+sred[3]+sred[4]+sred[5]+sred[6]+sred[7];
  float inv = rsqrtf(tot * (1.0f / D_MODEL) + EPSF);
  const float4* wp = (const float4*)(w + base);
  float4 w0 = wp[0], w1 = wp[1];
  float* o = out + (size_t)row * D_MODEL + base;
  *(float4*)o       = make_float4(a.x*inv*w0.x, a.y*inv*w0.y, a.z*inv*w0.z, a.w*inv*w0.w);
  *(float4*)(o + 4) = make_float4(b.x*inv*w1.x, b.y*inv*w1.y, b.z*inv*w1.z, b.w*inv*w1.w);
}

// ------------------------------------------- Generic WMMA GEMM: C = A*B^T (+R)
// A: (M,K) row-major fp32, B: (N,K) row-major fp32, C: (M,N) row-major fp32.
__global__ __launch_bounds__(256) void gemm_kernel(
    const float* __restrict__ A, int lda,
    const float* __restrict__ B, int ldb,
    float* __restrict__ C, int ldc,
    const float* __restrict__ R,
    int M, int N, int K) {
  __shared__ unsigned short lA[128 * 40];
  __shared__ unsigned short lB[128 * 40];
  int m0 = blockIdx.y * 128, n0 = blockIdx.x * 128;
  int tid = threadIdx.x, wid = tid >> 5, lane = tid & 31;
  int wm = wid >> 1, wn = wid & 1;
  int g = lane >> 4, nl = lane & 15;
  v8f acc[2][4] = {};
  for (int kb = 0; kb < K; kb += 32) {
    #pragma unroll
    for (int t = tid; t < 512; t += 256) {
      int row = t >> 2, c8 = (t & 3) << 3;
      float4 f0 = make_float4(0,0,0,0), f1 = make_float4(0,0,0,0);
      int gr = m0 + row;
      if (gr < M) {
        const float4* src = (const float4*)(A + (size_t)gr * lda + kb + c8);
        f0 = src[0]; f1 = src[1];
      }
      *(uint4*)&lA[row*40 + c8] =
        make_uint4(pk2(f0.x,f0.y), pk2(f0.z,f0.w), pk2(f1.x,f1.y), pk2(f1.z,f1.w));
    }
    #pragma unroll
    for (int t = tid; t < 512; t += 256) {
      int row = t >> 2, c8 = (t & 3) << 3;
      float4 f0 = make_float4(0,0,0,0), f1 = make_float4(0,0,0,0);
      int gr = n0 + row;
      if (gr < N) {
        const float4* src = (const float4*)(B + (size_t)gr * ldb + kb + c8);
        f0 = src[0]; f1 = src[1];
      }
      *(uint4*)&lB[row*40 + c8] =
        make_uint4(pk2(f0.x,f0.y), pk2(f0.z,f0.w), pk2(f1.x,f1.y), pk2(f1.z,f1.w));
    }
    // prefetch next K slab while this one computes
    if (kb + 32 < K) {
      int pr = m0 + (tid >> 1);
      if (pr < M) __builtin_prefetch(A + (size_t)pr * lda + kb + 32, 0, 3);
      int br = n0 + (tid >> 1);
      if (br < N) __builtin_prefetch(B + (size_t)br * ldb + kb + 32, 0, 3);
    }
    __syncthreads();
    Frag a0, a1;
    ld_frag(lA, wm*32 + nl,      40, 0, a0);
    ld_frag(lA, wm*32 + 16 + nl, 40, 0, a1);
    #pragma unroll
    for (int ni = 0; ni < 4; ++ni) {
      Frag bf; ld_frag(lB, wn*64 + ni*16 + nl, 40, 0, bf);
      acc[0][ni] = wmma_bf16(a0, bf, acc[0][ni]);
      acc[1][ni] = wmma_bf16(a1, bf, acc[1][ni]);
    }
    __syncthreads();
  }
  #pragma unroll
  for (int mi = 0; mi < 2; ++mi)
    #pragma unroll
    for (int ni = 0; ni < 4; ++ni)
      #pragma unroll
      for (int j = 0; j < 8; ++j) {
        int m = m0 + wm*32 + mi*16 + j + 8*g;
        int n = n0 + wn*64 + ni*16 + nl;
        if (m < M && n < N) {
          float v = acc[mi][ni][j];
          if (R) v += R[(size_t)m * ldc + n];
          C[(size_t)m * ldc + n] = v;
        }
      }
}

// ---------------------------------------------------------------- RoPE (q,k)
__global__ void rope_kernel(float* __restrict__ q, float* __restrict__ k) {
  int s = blockIdx.x;
  float pos = (float)s;
  for (int idx = threadIdx.x; idx < NHEADS * 64; idx += blockDim.x) {
    int h = idx >> 6, i = idx & 63;
    // inv_freq = theta^(-2i/HD) = exp(-(2i/128)*ln(10000))
    float inv = __expf(-(float)(2 * i) * (1.0f / HDIM) * 9.210340371976184f);
    float ang = pos * inv;
    float c = cosf(ang), sn = sinf(ang);
    size_t off = (size_t)s * D_MODEL + h * HDIM + i;
    float q1 = q[off], q2 = q[off + 64];
    q[off] = q1 * c - q2 * sn; q[off + 64] = q2 * c + q1 * sn;
    float k1 = k[off], k2 = k[off + 64];
    k[off] = k1 * c - k2 * sn; k[off + 64] = k2 * c + k1 * sn;
  }
}

// ------------------------------------------------ Flash attention (causal)
// Block = (q-tile of 128 rows, head). 8 waves; wave owns 16 q rows.
__global__ __launch_bounds__(256) void flash_attn_kernel(
    const float* __restrict__ Q, const float* __restrict__ Kb,
    const float* __restrict__ Vb, float* __restrict__ ctx) {
  const float SC = 0.08838834764831845f;
  int qt = blockIdx.x, h = blockIdx.y;
  int tid = threadIdx.x, wid = tid >> 5, lane = tid & 31;
  int g = lane >> 4, nl = lane & 15;
  __shared__ unsigned short lK[128 * 136];   // K tile; later aliased as P stage
  __shared__ unsigned short lV[128 * 136];   // V^T tile: [d][key]

  // preload Q fragments (scaled by ATTN_SCALE, folded into bf16 A)
  Frag qf[4];
  {
    int qrow = qt * 128 + wid * 16 + nl;
    const float* qp = Q + (size_t)qrow * D_MODEL + h * HDIM;
    #pragma unroll
    for (int kk = 0; kk < 4; ++kk) {
      const float4* p0 = (const float4*)(qp + kk * 32 + g * 8);
      float4 f0 = p0[0], f1 = p0[1];
      const float4* p1 = (const float4*)(qp + kk * 32 + 16 + g * 8);
      float4 f2 = p1[0], f3 = p1[1];
      qf[kk].q[0] = make_uint4(pk2(f0.x*SC,f0.y*SC), pk2(f0.z*SC,f0.w*SC),
                               pk2(f1.x*SC,f1.y*SC), pk2(f1.z*SC,f1.w*SC));
      qf[kk].q[1] = make_uint4(pk2(f2.x*SC,f2.y*SC), pk2(f2.z*SC,f2.w*SC),
                               pk2(f3.x*SC,f3.y*SC), pk2(f3.z*SC,f3.w*SC));
    }
  }

  v8f oacc[8] = {};
  float mrow[8], lrow[8];
  #pragma unroll
  for (int j = 0; j < 8; ++j) { mrow[j] = -1e30f; lrow[j] = 0.f; }

  for (int kt = 0; kt <= qt; ++kt) {
    // stage K tile (row-major, bf16) and V tile (transposed) into LDS
    for (int t = tid; t < 128 * 16; t += 256) {
      int row = t >> 4, c8 = (t & 15) << 3;
      const float4* src =
        (const float4*)(Kb + (size_t)(kt*128 + row) * D_MODEL + h*HDIM + c8);
      float4 f0 = src[0], f1 = src[1];
      *(uint4*)&lK[row*136 + c8] =
        make_uint4(pk2(f0.x,f0.y), pk2(f0.z,f0.w), pk2(f1.x,f1.y), pk2(f1.z,f1.w));
    }
    for (int t = tid; t < 128 * 16; t += 256) {
      int row = t >> 4, c8 = (t & 15) << 3;   // row = key index, c8 = d offset
      const float4* src =
        (const float4*)(Vb + (size_t)(kt*128 + row) * D_MODEL + h*HDIM + c8);
      float4 f0 = src[0], f1 = src[1];
      float tmp[8] = {f0.x,f0.y,f0.z,f0.w,f1.x,f1.y,f1.z,f1.w};
      #pragma unroll
      for (int i = 0; i < 8; ++i)
        lV[(size_t)(c8 + i) * 136 + row] = bf16bits(tmp[i]);
    }
    __syncthreads();

    // S = (scale*Q) K^T : 8 n-tiles x 4 k-steps
    v8f sacc[8] = {};
    #pragma unroll
    for (int kk = 0; kk < 4; ++kk)
      #pragma unroll
      for (int nt = 0; nt < 8; ++nt) {
        Frag kf; ld_frag(lK, nt*16 + nl, 136, kk*32, kf);
        sacc[nt] = wmma_bf16(qf[kk], kf, sacc[nt]);
      }
    __syncthreads();   // all waves done reading lK before it becomes P stage

    // online softmax (lane holds col nl of each n-tile, rows j+8g)
    bool diag = (kt == qt);
    #pragma unroll
    for (int j = 0; j < 8; ++j) {
      int qglob = qt*128 + wid*16 + j + 8*g;
      float mcur = -1e30f;
      #pragma unroll
      for (int nt = 0; nt < 8; ++nt) {
        if (diag) {
          int kglob = kt*128 + nt*16 + nl;
          if (kglob > qglob) sacc[nt][j] = -1e30f;
        }
        mcur = fmaxf(mcur, sacc[nt][j]);
      }
      #pragma unroll
      for (int msk = 1; msk < 16; msk <<= 1)
        mcur = fmaxf(mcur, __shfl_xor(mcur, msk, 32));
      float mnew = fmaxf(mrow[j], mcur);
      float alpha = __expf(mrow[j] - mnew);
      float ls = 0.f;
      #pragma unroll
      for (int nt = 0; nt < 8; ++nt) {
        float p = __expf(sacc[nt][j] - mnew);
        sacc[nt][j] = p;
        ls += p;
      }
      #pragma unroll
      for (int msk = 1; msk < 16; msk <<= 1)
        ls += __shfl_xor(ls, msk, 32);
      lrow[j] = lrow[j] * alpha + ls;
      mrow[j] = mnew;
      #pragma unroll
      for (int dt = 0; dt < 8; ++dt) oacc[dt][j] *= alpha;
    }

    // stage P (C-layout -> row-major bf16) into this wave's 16-row slice of lK
    #pragma unroll
    for (int j = 0; j < 8; ++j) {
      int r = j + 8*g;
      #pragma unroll
      for (int nt = 0; nt < 8; ++nt)
        lK[(size_t)(wid*16 + r) * 136 + nt*16 + nl] = bf16bits(sacc[nt][j]);
    }
    __syncthreads();

    // O += P * V  (A = P 16x128 keys, B^T = V^T rows)
    #pragma unroll
    for (int kk = 0; kk < 4; ++kk) {
      Frag pf; ld_frag(lK + (size_t)(wid*16) * 136, nl, 136, kk*32, pf);
      #pragma unroll
      for (int dt = 0; dt < 8; ++dt) {
        Frag vf; ld_frag(lV, dt*16 + nl, 136, kk*32, vf);
        oacc[dt] = wmma_bf16(pf, vf, oacc[dt]);
      }
    }
    __syncthreads();   // before next tile overwrites lK/lV
  }

  #pragma unroll
  for (int j = 0; j < 8; ++j) {
    int m = qt*128 + wid*16 + j + 8*g;
    float inv = 1.f / lrow[j];
    #pragma unroll
    for (int dt = 0; dt < 8; ++dt)
      ctx[(size_t)m * D_MODEL + h*HDIM + dt*16 + nl] = oacc[dt][j] * inv;
  }
}

// ------------------------------------------------------------ router top-k
__global__ void topk_kernel(const float* __restrict__ logits,
                            int* __restrict__ topi, float* __restrict__ topw,
                            int* __restrict__ counts) {
  int t = blockIdx.x;
  __shared__ float sm[NEXPERTS];
  int e = threadIdx.x;                       // blockDim = 64
  sm[e] = logits[t * NEXPERTS + e];
  __syncthreads();
  if (e == 0) {
    float mx = sm[0];
    for (int i = 1; i < NEXPERTS; ++i) mx = fmaxf(mx, sm[i]);
    float s = 0.f;
    for (int i = 0; i < NEXPERTS; ++i) { sm[i] = __expf(sm[i] - mx); s += sm[i]; }
    float invs = 1.f / s;
    for (int slot = 0; slot < NTOPK; ++slot) {
      int bi = 0; float bv = -1.f;
      for (int i = 0; i < NEXPERTS; ++i)
        if (sm[i] > bv) { bv = sm[i]; bi = i; }
      topi[t * NTOPK + slot] = bi;
      topw[t * NTOPK + slot] = bv * invs;
      sm[bi] = -1.f;
      atomicAdd(&counts[bi], 1);
    }
  }
}

__global__ void zero_ints_kernel(int* p, int n) {
  int i = blockIdx.x * blockDim.x + threadIdx.x;
  if (i < n) p[i] = 0;
}

__global__ void scan_kernel(const int* __restrict__ counts,
                            int* __restrict__ offs, int* __restrict__ fill) {
  if (threadIdx.x == 0) {
    int s = 0;
    for (int e = 0; e < NEXPERTS; ++e) { offs[e] = s; s += counts[e]; fill[e] = 0; }
  }
}

__global__ void scatter_kernel(const int* __restrict__ topi,
                               const float* __restrict__ topw,
                               const int* __restrict__ offs,
                               int* __restrict__ fill,
                               int* __restrict__ rowlist,
                               float* __restrict__ wlist) {
  int idx = blockIdx.x * blockDim.x + threadIdx.x;
  if (idx >= SEQ * NTOPK) return;
  int e = topi[idx];
  int pos = atomicAdd(&fill[e], 1);
  int row = offs[e] + pos;
  rowlist[row] = idx / NTOPK;
  wlist[row] = topw[idx];
}

// ------------------------- MoE gate+up fused gathered GEMM -> act = silu(g)*u
__global__ __launch_bounds__(256) void moe_gateup_kernel(
    const float* __restrict__ Hb, const float* __restrict__ gate_w,
    const float* __restrict__ up_w,
    const int* __restrict__ counts, const int* __restrict__ offs,
    const int* __restrict__ rowlist, float* __restrict__ act) {
  int e = blockIdx.z;
  int cnt = counts[e];
  int m0 = blockIdx.y * 128;
  if (m0 >= cnt) return;
  int base = offs[e];
  int n0 = blockIdx.x * 128;
  const float* Bg = gate_w + (size_t)e * FFDIM * D_MODEL;
  const float* Bu = up_w   + (size_t)e * FFDIM * D_MODEL;
  __shared__ unsigned short lA[128*40], lG[128*40], lU[128*40];
  int tid = threadIdx.x, wid = tid >> 5, lane = tid & 31;
  int wm = wid >> 1, wn = wid & 1;
  int g = lane >> 4, nl = lane & 15;
  v8f accg[2][4] = {}, accu[2][4] = {};
  for (int kb = 0; kb < D_MODEL; kb += 32) {
    #pragma unroll
    for (int t = tid; t < 512; t += 256) {
      int row = t >> 2, c8 = (t & 3) << 3;
      float4 f0 = make_float4(0,0,0,0), f1 = make_float4(0,0,0,0);
      int gr = m0 + row;
      if (gr < cnt) {
        int ar = rowlist[base + gr];
        const float4* src = (const float4*)(Hb + (size_t)ar * D_MODEL + kb + c8);
        f0 = src[0]; f1 = src[1];
      }
      *(uint4*)&lA[row*40 + c8] =
        make_uint4(pk2(f0.x,f0.y), pk2(f0.z,f0.w), pk2(f1.x,f1.y), pk2(f1.z,f1.w));
    }
    #pragma unroll
    for (int t = tid; t < 512; t += 256) {
      int row = t >> 2, c8 = (t & 3) << 3;
      const float4* sg = (const float4*)(Bg + (size_t)(n0+row) * D_MODEL + kb + c8);
      float4 f0 = sg[0], f1 = sg[1];
      *(uint4*)&lG[row*40 + c8] =
        make_uint4(pk2(f0.x,f0.y), pk2(f0.z,f0.w), pk2(f1.x,f1.y), pk2(f1.z,f1.w));
      const float4* su = (const float4*)(Bu + (size_t)(n0+row) * D_MODEL + kb + c8);
      float4 u0 = su[0], u1 = su[1];
      *(uint4*)&lU[row*40 + c8] =
        make_uint4(pk2(u0.x,u0.y), pk2(u0.z,u0.w), pk2(u1.x,u1.y), pk2(u1.z,u1.w));
    }
    __syncthreads();
    Frag a0, a1;
    ld_frag(lA, wm*32 + nl,      40, 0, a0);
    ld_frag(lA, wm*32 + 16 + nl, 40, 0, a1);
    #pragma unroll
    for (int ni = 0; ni < 4; ++ni) {
      Frag bg; ld_frag(lG, wn*64 + ni*16 + nl, 40, 0, bg);
      accg[0][ni] = wmma_bf16(a0, bg, accg[0][ni]);
      accg[1][ni] = wmma_bf16(a1, bg, accg[1][ni]);
      Frag bu; ld_frag(lU, wn*64 + ni*16 + nl, 40, 0, bu);
      accu[0][ni] = wmma_bf16(a0, bu, accu[0][ni]);
      accu[1][ni] = wmma_bf16(a1, bu, accu[1][ni]);
    }
    __syncthreads();
  }
  #pragma unroll
  for (int mi = 0; mi < 2; ++mi)
    #pragma unroll
    for (int ni = 0; ni < 4; ++ni)
      #pragma unroll
      for (int j = 0; j < 8; ++j) {
        int m = m0 + wm*32 + mi*16 + j + 8*g;
        if (m < cnt) {
          int n = n0 + wn*64 + ni*16 + nl;
          float gv = accg[mi][ni][j], uv = accu[mi][ni][j];
          float s = gv / (1.f + __expf(-gv));      // silu
          act[(size_t)(base + m) * FFDIM + n] = s * uv;
        }
      }
}

// --------------------------- MoE down GEMM: out[token] += w * (act @ down^T)
__global__ __launch_bounds__(256) void moe_down_kernel(
    const float* __restrict__ act, const float* __restrict__ down_w,
    const int* __restrict__ counts, const int* __restrict__ offs,
    const int* __restrict__ rowlist, const float* __restrict__ wlist,
    float* __restrict__ out) {
  int e = blockIdx.z;
  int cnt = counts[e];
  int m0 = blockIdx.y * 128;
  if (m0 >= cnt) return;
  int base = offs[e];
  int n0 = blockIdx.x * 128;
  const float* B = down_w + (size_t)e * D_MODEL * FFDIM;
  __shared__ unsigned short lA[128*40], lB[128*40];
  int tid = threadIdx.x, wid = tid >> 5, lane = tid & 31;
  int wm = wid >> 1, wn = wid & 1;
  int g = lane >> 4, nl = lane & 15;
  v8f acc[2][4] = {};
  for (int kb = 0; kb < FFDIM; kb += 32) {
    #pragma unroll
    for (int t = tid; t < 512; t += 256) {
      int row = t >> 2, c8 = (t & 3) << 3;
      float4 f0 = make_float4(0,0,0,0), f1 = make_float4(0,0,0,0);
      int gr = m0 + row;
      if (gr < cnt) {
        const float4* src = (const float4*)(act + (size_t)(base+gr) * FFDIM + kb + c8);
        f0 = src[0]; f1 = src[1];
      }
      *(uint4*)&lA[row*40 + c8] =
        make_uint4(pk2(f0.x,f0.y), pk2(f0.z,f0.w), pk2(f1.x,f1.y), pk2(f1.z,f1.w));
    }
    #pragma unroll
    for (int t = tid; t < 512; t += 256) {
      int row = t >> 2, c8 = (t & 3) << 3;
      const float4* src = (const float4*)(B + (size_t)(n0+row) * FFDIM + kb + c8);
      float4 f0 = src[0], f1 = src[1];
      *(uint4*)&lB[row*40 + c8] =
        make_uint4(pk2(f0.x,f0.y), pk2(f0.z,f0.w), pk2(f1.x,f1.y), pk2(f1.z,f1.w));
    }
    __syncthreads();
    Frag a0, a1;
    ld_frag(lA, wm*32 + nl,      40, 0, a0);
    ld_frag(lA, wm*32 + 16 + nl, 40, 0, a1);
    #pragma unroll
    for (int ni = 0; ni < 4; ++ni) {
      Frag bf; ld_frag(lB, wn*64 + ni*16 + nl, 40, 0, bf);
      acc[0][ni] = wmma_bf16(a0, bf, acc[0][ni]);
      acc[1][ni] = wmma_bf16(a1, bf, acc[1][ni]);
    }
    __syncthreads();
  }
  #pragma unroll
  for (int mi = 0; mi < 2; ++mi)
    #pragma unroll
    for (int ni = 0; ni < 4; ++ni)
      #pragma unroll
      for (int j = 0; j < 8; ++j) {
        int m = m0 + wm*32 + mi*16 + j + 8*g;
        if (m < cnt) {
          int tok = rowlist[base + m];
          float wgt = wlist[base + m];
          int n = n0 + wn*64 + ni*16 + nl;
          atomicAdd(&out[(size_t)tok * D_MODEL + n], wgt * acc[mi][ni][j]);
        }
      }
}

// ------------------------------------------------------------------ driver
extern "C" void kernel_launch(void* const* d_in, const int* in_sizes, int n_in,
                              void* d_out, int out_size, void* d_ws, size_t ws_size,
                              hipStream_t stream) {
  const float* x         = (const float*)d_in[0];
  const float* in_ln_w   = (const float*)d_in[1];
  const float* q_w       = (const float*)d_in[2];
  const float* q_norm_w  = (const float*)d_in[3];
  const float* k_w       = (const float*)d_in[4];
  const float* k_norm_w  = (const float*)d_in[5];
  const float* v_w       = (const float*)d_in[6];
  const float* o_w       = (const float*)d_in[7];
  const float* post_ln_w = (const float*)d_in[8];
  const float* router_w  = (const float*)d_in[9];
  const float* gate_w    = (const float*)d_in[10];
  const float* up_w      = (const float*)d_in[11];
  const float* down_w    = (const float*)d_in[12];
  float* out = (float*)d_out;

  const size_t SD = (size_t)SEQ * D_MODEL;
  float* ws0 = (float*)d_ws;
  float* xn  = ws0;            // later reused as ctx
  float* qb  = ws0 + SD;       // later reused as x2
  float* kb  = ws0 + 2 * SD;   // later reused as h
  float* vb  = ws0 + 3 * SD;   // later reused as logits
  float* act = ws0 + 4 * SD;                       // 8192 x FFDIM
  float* topw  = act + (size_t)SEQ * NTOPK * FFDIM;
  float* wlist = topw + SEQ * NTOPK;
  int* ipool   = (int*)(wlist + SEQ * NTOPK);
  int* topi    = ipool;                 // 8192
  int* counts  = ipool + SEQ * NTOPK;   // 64
  int* offs    = counts + NEXPERTS;     // 64
  int* fill    = offs + NEXPERTS;       // 64
  int* rowlist = fill + NEXPERTS;       // 8192

  float* ctx    = xn;
  float* x2     = qb;
  float* hb     = kb;
  float* logits = vb;

  // 1) xn = rmsnorm(x)
  rmsnorm_kernel<<<SEQ, 256, 0, stream>>>(x, in_ln_w, xn);
  // 2) q,k,v projections
  gemm_kernel<<<dim3(16, 8), 256, 0, stream>>>(xn, D_MODEL, q_w, D_MODEL,
      qb, D_MODEL, nullptr, SEQ, D_MODEL, D_MODEL);
  gemm_kernel<<<dim3(16, 8), 256, 0, stream>>>(xn, D_MODEL, k_w, D_MODEL,
      kb, D_MODEL, nullptr, SEQ, D_MODEL, D_MODEL);
  gemm_kernel<<<dim3(16, 8), 256, 0, stream>>>(xn, D_MODEL, v_w, D_MODEL,
      vb, D_MODEL, nullptr, SEQ, D_MODEL, D_MODEL);
  // 3) q/k norms (in place) + RoPE
  rmsnorm_kernel<<<SEQ, 256, 0, stream>>>(qb, q_norm_w, qb);
  rmsnorm_kernel<<<SEQ, 256, 0, stream>>>(kb, k_norm_w, kb);
  rope_kernel<<<SEQ, 256, 0, stream>>>(qb, kb);
  // 4) flash attention -> ctx (reuses xn)
  flash_attn_kernel<<<dim3(8, NHEADS), 256, 0, stream>>>(qb, kb, vb, ctx);
  // 5) x2 = x + ctx @ o_w^T
  gemm_kernel<<<dim3(16, 8), 256, 0, stream>>>(ctx, D_MODEL, o_w, D_MODEL,
      x2, D_MODEL, x, SEQ, D_MODEL, D_MODEL);
  // 6) h = rmsnorm(x2); router logits
  rmsnorm_kernel<<<SEQ, 256, 0, stream>>>(x2, post_ln_w, hb);
  gemm_kernel<<<dim3(1, 8), 256, 0, stream>>>(hb, D_MODEL, router_w, D_MODEL,
      logits, NEXPERTS, nullptr, SEQ, NEXPERTS, D_MODEL);
  // 7) routing: top-8, CSR build
  zero_ints_kernel<<<1, NEXPERTS, 0, stream>>>(counts, NEXPERTS);
  topk_kernel<<<SEQ, NEXPERTS, 0, stream>>>(logits, topi, topw, counts);
  scan_kernel<<<1, 1, 0, stream>>>(counts, offs, fill);
  scatter_kernel<<<(SEQ * NTOPK + 255) / 256, 256, 0, stream>>>(
      topi, topw, offs, fill, rowlist, wlist);
  // 8) out = x2, then scatter-add MoE
  hipMemcpyAsync(out, x2, SD * sizeof(float), hipMemcpyDeviceToDevice, stream);
  moe_gateup_kernel<<<dim3(FFDIM / 128, 8, NEXPERTS), 256, 0, stream>>>(
      hb, gate_w, up_w, counts, offs, rowlist, act);
  moe_down_kernel<<<dim3(D_MODEL / 128, 8, NEXPERTS), 256, 0, stream>>>(
      act, down_w, counts, offs, rowlist, wlist, out);
}
